// SpatialCTX_55018531062716
// MI455X (gfx1250) — matrix operand order repaired and checked by reading, not compiled, not run on hardware
//
#include <hip/hip_runtime.h>

typedef __attribute__((ext_vector_type(16))) _Float16 v16h;
typedef __attribute__((ext_vector_type(8)))  _Float16 v8h;
typedef __attribute__((ext_vector_type(8)))  float    v8f;
typedef __attribute__((ext_vector_type(4)))  float    v4f;

#define KPS     39
#define D_IN    78
#define D_HID   195
#define D_IN_P  96        // K padding for GEMM1 (3 x 32)
#define D_HID_P 224       // K padding for GEMM2 (7 x 32)
#define H_STR   232       // LDS row stride (halves) for h buffer, 16B aligned
#define J_STR   80        // LDS row stride (floats) for residual buffer
#define M_TILE  64
#define NROWS   (256L * 2048L)
#define NT1     13        // N tiles GEMM1 (208 >= 195)
#define NT2     5         // N tiles GEMM2 (80 >= 78)
#define W1P_ELEMS (208 * D_IN_P)
#define W2P_ELEMS (80 * D_HID_P)
#define W1P_BYTES (W1P_ELEMS * 2)          // 39936
#define W2P_BYTES (W2P_ELEMS * 2)          // 35840
#define P1_OFF    (W1P_BYTES + W2P_BYTES)  // 75776 (16B aligned)
#define P2_OFF    (P1_OFF + 208 * 16)      // 79104

// ---------------------------------------------------------------------------
// Prep: pack weights to padded f16 in WMMA-B-friendly layout, and pack
// (bias, alpha, beta, 0) per output column (zeros in pad columns so padded
// lanes flow through bias/PReLU/LN as exact zero -> no predication needed).
// ---------------------------------------------------------------------------
__global__ __launch_bounds__(256) void prep_weights(
    const float* __restrict__ W1, const float* __restrict__ W2,
    const float* __restrict__ b1, const float* __restrict__ alpha1,
    const float* __restrict__ beta1,
    const float* __restrict__ b2, const float* __restrict__ alpha2,
    const float* __restrict__ beta2,
    _Float16* __restrict__ W1p, _Float16* __restrict__ W2p,
    float* __restrict__ P1, float* __restrict__ P2)
{
    int i = blockIdx.x * blockDim.x + threadIdx.x;
    if (i < W1P_ELEMS) {
        int n = i / D_IN_P, k = i % D_IN_P;
        W1p[i] = (n < D_HID && k < D_IN) ? (_Float16)W1[n * D_IN + k] : (_Float16)0.0f;
    }
    if (i < W2P_ELEMS) {
        int n = i / D_HID_P, k = i % D_HID_P;
        W2p[i] = (n < D_IN && k < D_HID) ? (_Float16)W2[n * D_HID + k] : (_Float16)0.0f;
    }
    if (i < 208) {
        bool v = i < D_HID;
        P1[i * 4 + 0] = v ? b1[i]     : 0.0f;
        P1[i * 4 + 1] = v ? alpha1[i] : 0.0f;
        P1[i * 4 + 2] = v ? beta1[i]  : 0.0f;
        P1[i * 4 + 3] = 0.0f;
    }
    if (i < 80) {
        bool v = i < D_IN;
        P2[i * 4 + 0] = v ? b2[i]     : 0.0f;
        P2[i * 4 + 1] = v ? alpha2[i] : 0.0f;
        P2[i * 4 + 2] = v ? beta2[i]  : 0.0f;
        P2[i * 4 + 3] = 0.0f;
    }
}

// v16h A-fragment (16-bit A 16x32 layout): K = hi*8 + {0..7} and hi*8 + {16..23}
__device__ __forceinline__ v16h load_a_frag(const _Float16* p) {
    v8h lo = *(const v8h*)p;
    v8h hv = *(const v8h*)(p + 16);
    return __builtin_shufflevector(lo, hv, 0,1,2,3,4,5,6,7,8,9,10,11,12,13,14,15);
}

// v16h B-fragment (32x16 layout): K = hi*8 + {0..7} and hi*8 + {16..23}
__device__ __forceinline__ v16h load_b_frag(const _Float16* p) {
    v8h lo = *(const v8h*)p;
    v8h hv = *(const v8h*)(p + 16);
    return __builtin_shufflevector(lo, hv, 0,1,2,3,4,5,6,7,8,9,10,11,12,13,14,15);
}

// ---------------------------------------------------------------------------
// Fused: joined = cat(x,y); h = LN(PReLU(joined @ W1^T + b1));
//        o = LN(PReLU(h @ W2^T + b2)) + joined; split outputs.
// 64 rows / workgroup, 4 waves, each wave owns 16 rows end-to-end:
// no barriers at all. Residual f32 copy rides the async-to-LDS path
// (ASYNCcnt) in the background behind both GEMMs.
// ---------------------------------------------------------------------------
__global__ __launch_bounds__(128) void mlp_fused(
    const float* __restrict__ x, const float* __restrict__ y,
    const _Float16* __restrict__ W1p, const _Float16* __restrict__ W2p,
    const float* __restrict__ P1, const float* __restrict__ P2,
    const float* __restrict__ a1p, const float* __restrict__ a2p,
    float* __restrict__ out)
{
    __shared__ alignas(16) _Float16 jf16[M_TILE * D_IN_P];  // 12288 B
    __shared__ alignas(16) float    jf32[M_TILE * J_STR];   // 20480 B
    __shared__ alignas(16) _Float16 hbuf[M_TILE * H_STR];   // 29696 B

    const int tid  = threadIdx.x;
    const int lane = tid & 31;
    const int wave = tid >> 5;
    const int rw   = wave * 16;        // this wave's row base within tile
    const int lr   = lane & 15;        // N/row selector within fragment
    const int hi   = (lane >> 4) & 1;  // half-wave selector
    const long rowBaseG = (long)blockIdx.x * M_TILE;

    // ---- Kick off async residual copy (global f32 -> LDS), ASYNCcnt path ----
    for (int rr = 0; rr < 16; ++rr) {
        const int  r  = rw + rr;
        const long rg = rowBaseG + r;
        #pragma unroll
        for (int cc = 0; cc < 3; ++cc) {
            int c = cc * 32 + lane;
            if (c < D_IN) {
                const float* g = (c < KPS) ? (x + rg * KPS + c)
                                           : (y + rg * KPS + (c - KPS));
                unsigned ldsoff = (unsigned)(unsigned long long)&jf32[r * J_STR + c];
                asm volatile("global_load_async_to_lds_b32 %0, %1, off"
                             :: "v"(ldsoff), "v"(g) : "memory");
            }
        }
    }

    // ---- Stage joined tile as f16 (K padded to 96 with zeros) ----
    for (int rr = 0; rr < 16; ++rr) {
        const int  r  = rw + rr;
        const long rg = rowBaseG + r;
        #pragma unroll
        for (int cc = 0; cc < 3; ++cc) {
            int c = cc * 32 + lane;
            float v = 0.0f;
            if (c < KPS)       v = x[rg * KPS + c];
            else if (c < D_IN) v = y[rg * KPS + (c - KPS)];
            jf16[r * D_IN_P + c] = (_Float16)v;
        }
    }

    const float a1 = a1p[0];
    const float a2 = a2p[0];

    // ---- GEMM1: [16 x 78] x [78 x 195], K padded to 96 ----
    v16h A1[3];
    #pragma unroll
    for (int kt = 0; kt < 3; ++kt)
        A1[kt] = load_a_frag(&jf16[(rw + lr) * D_IN_P + kt * 32 + hi * 8]);

    v8f acc[NT1];
    #pragma unroll
    for (int nt = 0; nt < NT1; ++nt) {
        v8f z = {};
        acc[nt] = z;
        #pragma unroll
        for (int kt = 0; kt < 3; ++kt) {
            v16h B = load_b_frag(&W1p[(nt * 16 + lr) * D_IN_P + kt * 32 + hi * 8]);
            acc[nt] = __builtin_amdgcn_wmma_f32_16x16x32_f16(
                false, A1[kt], false, B, (short)0, acc[nt], false, false);
        }
    }

    // ---- bias + PReLU + LayerNorm(195); pad columns are exact zeros ----
    float s[8], q[8];
    #pragma unroll
    for (int r = 0; r < 8; ++r) { s[r] = 0.0f; q[r] = 0.0f; }
    #pragma unroll
    for (int nt = 0; nt < NT1; ++nt) {
        v4f pb = *(const v4f*)(P1 + (nt * 16 + lr) * 4);
        #pragma unroll
        for (int r = 0; r < 8; ++r) {
            float v = acc[nt][r] + pb[0];
            v = (v >= 0.0f) ? v : a1 * v;
            acc[nt][r] = v;
            s[r] += v; q[r] += v * v;
        }
    }
    #pragma unroll
    for (int m = 1; m < 16; m <<= 1) {
        #pragma unroll
        for (int r = 0; r < 8; ++r) {
            s[r] += __shfl_xor(s[r], m, 32);
            q[r] += __shfl_xor(q[r], m, 32);
        }
    }
    float mean[8], rstd[8];
    #pragma unroll
    for (int r = 0; r < 8; ++r) {
        float mu  = s[r] * (1.0f / 195.0f);
        float var = q[r] * (1.0f / 195.0f) - mu * mu;
        mean[r] = mu;
        rstd[r] = rsqrtf(var + 1e-5f);
    }
    // normalize, scale/shift (alpha=beta=0 in pad cols -> stores exact 0)
    #pragma unroll
    for (int nt = 0; nt < NT1; ++nt) {
        v4f pb = *(const v4f*)(P1 + (nt * 16 + lr) * 4);
        #pragma unroll
        for (int r = 0; r < 8; ++r) {
            float v = (acc[nt][r] - mean[r]) * rstd[r] * pb[1] + pb[2];
            hbuf[(rw + hi * 8 + r) * H_STR + (nt * 16 + lr)] = (_Float16)v;
        }
    }
    // zero K-pad region 208..223 (read by GEMM2's last K tile)
    #pragma unroll
    for (int r = 0; r < 8; ++r)
        hbuf[(rw + hi * 8 + r) * H_STR + 208 + lr] = (_Float16)0.0f;
    // No barrier: this wave reads back only its own rows; per-wave LDS ops
    // execute in order.

    // ---- GEMM2: [16 x 195] x [195 x 78], K padded to 224 ----
    v8f acc2[NT2];
    #pragma unroll
    for (int nt = 0; nt < NT2; ++nt) { v8f z = {}; acc2[nt] = z; }
    #pragma unroll
    for (int kt = 0; kt < 7; ++kt) {
        v16h A = load_a_frag(&hbuf[(rw + lr) * H_STR + kt * 32 + hi * 8]);
        #pragma unroll
        for (int nt = 0; nt < NT2; ++nt) {
            v16h B = load_b_frag(&W2p[(nt * 16 + lr) * D_HID_P + kt * 32 + hi * 8]);
            acc2[nt] = __builtin_amdgcn_wmma_f32_16x16x32_f16(
                false, A, false, B, (short)0, acc2[nt], false, false);
        }
    }

    // ---- bias + PReLU + LayerNorm(78) ----
    float s2[8], q2[8];
    #pragma unroll
    for (int r = 0; r < 8; ++r) { s2[r] = 0.0f; q2[r] = 0.0f; }
    #pragma unroll
    for (int nt = 0; nt < NT2; ++nt) {
        v4f pb = *(const v4f*)(P2 + (nt * 16 + lr) * 4);
        #pragma unroll
        for (int r = 0; r < 8; ++r) {
            float v = acc2[nt][r] + pb[0];
            v = (v >= 0.0f) ? v : a2 * v;
            acc2[nt][r] = v;
            s2[r] += v; q2[r] += v * v;
        }
    }
    #pragma unroll
    for (int m = 1; m < 16; m <<= 1) {
        #pragma unroll
        for (int r = 0; r < 8; ++r) {
            s2[r] += __shfl_xor(s2[r], m, 32);
            q2[r] += __shfl_xor(q2[r], m, 32);
        }
    }
    float mean2[8], rstd2[8];
    #pragma unroll
    for (int r = 0; r < 8; ++r) {
        float mu  = s2[r] * (1.0f / 78.0f);
        float var = q2[r] * (1.0f / 78.0f) - mu * mu;
        mean2[r] = mu;
        rstd2[r] = rsqrtf(var + 1e-5f);
    }

    // ---- wait for async residual, then add + split nontemporal store ----
    asm volatile("s_wait_asynccnt 0x0" ::: "memory");
    #pragma unroll
    for (int nt = 0; nt < NT2; ++nt) {
        int col = nt * 16 + lr;
        v4f pb = *(const v4f*)(P2 + col * 4);
        if (col < D_IN) {
            #pragma unroll
            for (int r = 0; r < 8; ++r) {
                int  row = rw + hi * 8 + r;
                long rg  = rowBaseG + row;
                float res = jf32[row * J_STR + col];
                float v = (acc2[nt][r] - mean2[r]) * rstd2[r] * pb[1] + pb[2] + res;
                if (col < KPS)
                    __builtin_nontemporal_store(v, &out[rg * KPS + col]);
                else
                    __builtin_nontemporal_store(v, &out[NROWS * KPS + rg * KPS + (col - KPS)]);
            }
        }
    }
}

extern "C" void kernel_launch(void* const* d_in, const int* in_sizes, int n_in,
                              void* d_out, int out_size, void* d_ws, size_t ws_size,
                              hipStream_t stream) {
    const float* x      = (const float*)d_in[0];
    const float* y      = (const float*)d_in[1];
    const float* W1     = (const float*)d_in[2];
    const float* b1     = (const float*)d_in[3];
    const float* W2     = (const float*)d_in[4];
    const float* b2     = (const float*)d_in[5];
    const float* a1     = (const float*)d_in[6];
    const float* a2     = (const float*)d_in[7];
    const float* alpha1 = (const float*)d_in[8];
    const float* beta1  = (const float*)d_in[9];
    const float* alpha2 = (const float*)d_in[10];
    const float* beta2  = (const float*)d_in[11];
    float* out = (float*)d_out;

    _Float16* W1p = (_Float16*)d_ws;
    _Float16* W2p = (_Float16*)((char*)d_ws + W1P_BYTES);
    float*    P1  = (float*)((char*)d_ws + P1_OFF);
    float*    P2  = (float*)((char*)d_ws + P2_OFF);

    int prep_n = (W1P_ELEMS > W2P_ELEMS) ? W1P_ELEMS : W2P_ELEMS;
    prep_weights<<<(prep_n + 255) / 256, 256, 0, stream>>>(
        W1, W2, b1, alpha1, beta1, b2, alpha2, beta2, W1p, W2p, P1, P2);

    long nblocks = NROWS / M_TILE;  // 8192
    mlp_fused<<<(int)nblocks, 128, 0, stream>>>(
        x, y, W1p, W2p, P1, P2, a1, a2, out);
}
